// Atom_Inter_Layer_62672162783752
// MI455X (gfx1250) — compile-verified, matrix operand order checked by reading, not run on hardware
//
#include <hip/hip_runtime.h>

typedef __attribute__((ext_vector_type(16))) __bf16 v16bf;
typedef __attribute__((ext_vector_type(4)))  __bf16 bf16x4;
typedef __attribute__((ext_vector_type(8)))  float  v8f;

static constexpr int NN   = 10000;
static constexpr int EE   = 320000;
static constexpr int DD   = 256;
static constexpr int EDIM = 64;

#define DEVI __device__ __forceinline__

// Load a 16x32 bf16 fragment where each lane's 16 k-values are contiguous in
// memory per the CDNA5 WMMA A/B per-lane layout (ISA 7.12.2):
//   lanes 0-15 : k = {0..7, 16..23} (+8*half handled by caller pointer)
//   lanes16-31 : k = {8..15, 24..31}
// Caller passes p = rowBase + k0 + 8*(lane>>4).
DEVI v16bf load_frag_rowk(const __bf16* __restrict__ p) {
  v16bf f;
#pragma unroll
  for (int v = 0; v < 8; ++v) {
    int k = (v < 4) ? (2 * v) : (16 + 2 * (v - 4));
    f[2 * v]     = p[k];
    f[2 * v + 1] = p[k + 1];
  }
  return f;
}

constexpr size_t mlp_smem(int K1, int HID) {
  const int K1P = K1 + 8;
  const int HP  = HID + 8;
  return (size_t)32 * K1P * 2      // sA  (bf16 input tile)
       + (size_t)32 * HP * 4       // sH  (fp32 hidden)
       + (size_t)32 * HP * 2       // sHb (bf16 activated hidden)
       + 256 * 4 * 2               // redS/redQ
       + 32 * 4 * 2;               // rowm/rowr
}

// Fused: [optional gather+concat] -> [optional LN] -> GEMM1(+b1)
//        -> [LN or pass] + SiLU -> GEMM2(+b2) -> out (bf16 or fp32)
// Block = 256 threads (8 waves), TM = 32 rows, full output width (256) owned
// by the block so row-wise LayerNorm stays on-chip.
template <int K1, int HID, bool EDGE_MODE, bool LN_IN, bool LN_HID, bool OUT_BF16>
__global__ __launch_bounds__(256)
void mlp_fused(const float* __restrict__ inp,      // plain mode input [rows,K1]
               const float* __restrict__ x,        // edge mode node feats [N,256]
               const float* __restrict__ eattr,    // edge mode [E,64]
               const int*   __restrict__ srcp,
               const int*   __restrict__ dstp,
               const float* __restrict__ lnin_g,
               const float* __restrict__ lnin_b,
               const __bf16* __restrict__ w1T,     // [HID][K1] bf16
               const float* __restrict__ b1,
               const float* __restrict__ g1,
               const float* __restrict__ be1,
               const __bf16* __restrict__ w2T,     // [256][HID] bf16
               const float* __restrict__ b2,
               void* __restrict__ outp,            // [rows,256]
               int rows) {
  constexpr int K1P = K1 + 8;
  constexpr int HP  = HID + 8;

  extern __shared__ char smem[];
  __bf16* sA   = (__bf16*)smem;
  float*  sH   = (float*)(smem + (size_t)32 * K1P * 2);
  __bf16* sHb  = (__bf16*)(smem + (size_t)32 * K1P * 2 + (size_t)32 * HP * 4);
  float*  redS = (float*)(smem + (size_t)32 * K1P * 2 + (size_t)32 * HP * 4 +
                          (size_t)32 * HP * 2);
  float*  redQ = redS + 256;
  float*  rowm = redQ + 256;
  float*  rowr = rowm + 32;

  const int tid  = threadIdx.x;
  const int r    = tid >> 3;   // row within tile, 8 threads per row
  const int sub  = tid & 7;
  const int row0 = blockIdx.x * 32;
  const int row  = row0 + r;
  const bool valid = row < rows;

  int si = 0, di = 0;
  if constexpr (EDGE_MODE) {
    if (valid) { si = srcp[row]; di = dstp[row]; }
  }

  // Vectorized fetch of 4 contiguous floats at f (f % 4 == 0; concat segment
  // boundaries 64/320 are multiples of 4 so a chunk never straddles).
  auto fetch4 = [&](int f) -> float4 {
    if constexpr (EDGE_MODE) {
      if (f < EDIM)           return *(const float4*)(eattr + (size_t)row * EDIM + f);
      else if (f < EDIM + DD) return *(const float4*)(x + (size_t)si * DD + (f - EDIM));
      else                    return *(const float4*)(x + (size_t)di * DD + (f - EDIM - DD));
    } else {
      return *(const float4*)(inp + (size_t)row * K1 + f);
    }
  };

  // ---- Phase 1: build bf16 input tile (with optional input LayerNorm) ----
  constexpr int NCH = K1 / 32;  // float4 chunks per sub-thread
  float mI = 0.f, rsI = 1.f;
  if constexpr (LN_IN) {
    float s = 0.f, q2 = 0.f;
    if (valid) {
#pragma unroll
      for (int j = 0; j < NCH; ++j) {
        int f = (sub + j * 8) * 4;
        float4 v = fetch4(f);
        s  += v.x + v.y + v.z + v.w;
        q2 += v.x * v.x + v.y * v.y + v.z * v.z + v.w * v.w;
      }
    }
    redS[tid] = s; redQ[tid] = q2;
    __syncthreads();
    if (tid < 32) {
      float ss = 0.f, qq = 0.f;
      for (int j = 0; j < 8; ++j) { ss += redS[tid * 8 + j]; qq += redQ[tid * 8 + j]; }
      float m   = ss * (1.f / K1);
      float var = qq * (1.f / K1) - m * m;
      rowm[tid] = m;
      rowr[tid] = rsqrtf(var + 1e-5f);
    }
    __syncthreads();
    mI = rowm[r]; rsI = rowr[r];
  }
#pragma unroll
  for (int j = 0; j < NCH; ++j) {
    int f = (sub + j * 8) * 4;
    float4 v;
    if (valid) v = fetch4(f);
    else       v = make_float4(0.f, 0.f, 0.f, 0.f);
    if constexpr (LN_IN) {
      float4 g = *(const float4*)(lnin_g + f);
      float4 b = *(const float4*)(lnin_b + f);
      v.x = (v.x - mI) * rsI * g.x + b.x;
      v.y = (v.y - mI) * rsI * g.y + b.y;
      v.z = (v.z - mI) * rsI * g.z + b.z;
      v.w = (v.w - mI) * rsI * g.w + b.w;
    }
    bf16x4 o;
    o[0] = (__bf16)v.x; o[1] = (__bf16)v.y; o[2] = (__bf16)v.z; o[3] = (__bf16)v.w;
    *(bf16x4*)(sA + (size_t)r * K1P + f) = o;
  }
  __syncthreads();

  // ---- Phase 2: GEMM1  [32,K1] x [K1,HID] -> sH ----
  const int wid = tid >> 5, lane = tid & 31;
  const int lm = lane & 15, lh = lane >> 4;
  constexpr int NT1 = HID / 16;
  for (int t = wid; t < 2 * NT1; t += 8) {
    int rt = t & 1, ct = t >> 1;
    if (t + 8 < 2 * NT1)
      __builtin_prefetch(w1T + (size_t)((ct + 4) * 16 + lm) * K1, 0, 0);
    v8f acc = {0.f, 0.f, 0.f, 0.f, 0.f, 0.f, 0.f, 0.f};
    const __bf16* ap = sA + (size_t)(rt * 16 + lm) * K1P + 8 * lh;
    const __bf16* bp = w1T + (size_t)(ct * 16 + lm) * K1 + 8 * lh;
    for (int k0 = 0; k0 < K1; k0 += 32) {
      v16bf fa = load_frag_rowk(ap + k0);
      v16bf fb = load_frag_rowk(bp + k0);
      acc = __builtin_amdgcn_wmma_f32_16x16x32_bf16(false, fa, false, fb,
                                                    (short)0, acc, false, false);
    }
    float bias = b1[ct * 16 + lm];
#pragma unroll
    for (int rr2 = 0; rr2 < 8; ++rr2) {
      int mrow = rt * 16 + rr2 + 8 * lh;
      sH[(size_t)mrow * HP + ct * 16 + lm] = acc[rr2] + bias;
    }
  }
  __syncthreads();

  // ---- Phase 3: (LN) + SiLU -> sHb bf16 ----
  constexpr int NCH2 = HID / 32;
  if constexpr (LN_HID) {
    float s = 0.f, q2 = 0.f;
#pragma unroll
    for (int j = 0; j < NCH2; ++j) {
      int c = (sub + j * 8) * 4;
      float4 h = *(const float4*)(sH + (size_t)r * HP + c);
      s  += h.x + h.y + h.z + h.w;
      q2 += h.x * h.x + h.y * h.y + h.z * h.z + h.w * h.w;
    }
    redS[tid] = s; redQ[tid] = q2;
    __syncthreads();
    if (tid < 32) {
      float ss = 0.f, qq = 0.f;
      for (int j = 0; j < 8; ++j) { ss += redS[tid * 8 + j]; qq += redQ[tid * 8 + j]; }
      float m   = ss * (1.f / HID);
      float var = qq * (1.f / HID) - m * m;
      rowm[tid] = m;
      rowr[tid] = rsqrtf(var + 1e-5f);
    }
    __syncthreads();
    float m2 = rowm[r], rs2 = rowr[r];
#pragma unroll
    for (int j = 0; j < NCH2; ++j) {
      int c = (sub + j * 8) * 4;
      float4 h = *(const float4*)(sH + (size_t)r * HP + c);
      float4 g = *(const float4*)(g1 + c);
      float4 b = *(const float4*)(be1 + c);
      h.x = (h.x - m2) * rs2 * g.x + b.x;
      h.y = (h.y - m2) * rs2 * g.y + b.y;
      h.z = (h.z - m2) * rs2 * g.z + b.z;
      h.w = (h.w - m2) * rs2 * g.w + b.w;
      bf16x4 o;
      o[0] = (__bf16)(h.x / (1.f + __expf(-h.x)));
      o[1] = (__bf16)(h.y / (1.f + __expf(-h.y)));
      o[2] = (__bf16)(h.z / (1.f + __expf(-h.z)));
      o[3] = (__bf16)(h.w / (1.f + __expf(-h.w)));
      *(bf16x4*)(sHb + (size_t)r * HP + c) = o;
    }
  } else {
#pragma unroll
    for (int j = 0; j < NCH2; ++j) {
      int c = (sub + j * 8) * 4;
      float4 h = *(const float4*)(sH + (size_t)r * HP + c);
      bf16x4 o;
      o[0] = (__bf16)(h.x / (1.f + __expf(-h.x)));
      o[1] = (__bf16)(h.y / (1.f + __expf(-h.y)));
      o[2] = (__bf16)(h.z / (1.f + __expf(-h.z)));
      o[3] = (__bf16)(h.w / (1.f + __expf(-h.w)));
      *(bf16x4*)(sHb + (size_t)r * HP + c) = o;
    }
  }
  __syncthreads();

  // ---- Phase 4: GEMM2  [32,HID] x [HID,256] -> global ----
  for (int t = wid; t < 32; t += 8) {
    int rt = t & 1, ct = t >> 1;
    if (t + 8 < 32)
      __builtin_prefetch(w2T + (size_t)((ct + 4) * 16 + lm) * HID, 0, 0);
    v8f acc = {0.f, 0.f, 0.f, 0.f, 0.f, 0.f, 0.f, 0.f};
    const __bf16* ap = sHb + (size_t)(rt * 16 + lm) * HP + 8 * lh;
    const __bf16* bp = w2T + (size_t)(ct * 16 + lm) * HID + 8 * lh;
    for (int k0 = 0; k0 < HID; k0 += 32) {
      v16bf fa = load_frag_rowk(ap + k0);
      v16bf fb = load_frag_rowk(bp + k0);
      acc = __builtin_amdgcn_wmma_f32_16x16x32_bf16(false, fa, false, fb,
                                                    (short)0, acc, false, false);
    }
    float bias = b2[ct * 16 + lm];
#pragma unroll
    for (int rr2 = 0; rr2 < 8; ++rr2) {
      int grow = row0 + rt * 16 + rr2 + 8 * lh;
      if (grow < rows) {
        size_t o = (size_t)grow * 256 + ct * 16 + lm;
        if constexpr (OUT_BF16) ((__bf16*)outp)[o] = (__bf16)(acc[rr2] + bias);
        else                    ((float*)outp)[o]  = acc[rr2] + bias;
      }
    }
  }
}

// ---- weight transpose + fp32->bf16 convert:  w[K][Nc] -> wT[Nc][K] ----
__global__ __launch_bounds__(256)
void convert_wT(const float* __restrict__ w, __bf16* __restrict__ wT, int K, int Nc) {
  int i = blockIdx.x * 256 + threadIdx.x;
  if (i >= K * Nc) return;
  int k = i / Nc, n = i % Nc;
  wT[(size_t)n * K + k] = (__bf16)w[i];
}

// ---- attention helpers ----
__global__ __launch_bounds__(256)
void init_attn(int* __restrict__ mx, float* __restrict__ den, float* __restrict__ acc) {
  int gid = blockIdx.x * 256 + threadIdx.x;
  if (gid < NN * 8) { mx[gid] = (int)0x807fffff; den[gid] = 0.f; }  // enc(-inf)
  if (gid < NN * 256) acc[gid] = 0.f;
}

__global__ __launch_bounds__(256)
void attn_score_max(const __bf16* __restrict__ kb, const float* __restrict__ qb,
                    const int* __restrict__ dstp, float* __restrict__ score,
                    int* __restrict__ mx) {
  int gid = blockIdx.x * 256 + threadIdx.x;
  if (gid >= EE * 8) return;
  int e = gid >> 3, h = gid & 7;
  int d = dstp[e];
  const __bf16* kp = kb + (size_t)e * 256 + h * 32;
  const float*  qp = qb + (size_t)d * 256 + h * 32;
  float s = 0.f;
#pragma unroll
  for (int c = 0; c < 32; ++c) s += (float)kp[c] * qp[c];
  s *= 0.17677669529663687f;  // 1/sqrt(32)
  score[gid] = s;
  int b = __float_as_int(s);
  b = (b >= 0) ? b : (b ^ 0x7fffffff);   // order-preserving int encoding
  atomicMax(&mx[d * 8 + h], b);
}

__global__ __launch_bounds__(256)
void attn_exp_den(const float* __restrict__ score, const int* __restrict__ dstp,
                  const int* __restrict__ mx, float* __restrict__ ebuf,
                  float* __restrict__ den) {
  int gid = blockIdx.x * 256 + threadIdx.x;
  if (gid >= EE * 8) return;
  int e = gid >> 3, h = gid & 7;
  int d = dstp[e];
  int bm = mx[d * 8 + h];
  float m = __int_as_float((bm >= 0) ? bm : (bm ^ 0x7fffffff));
  float ex = __expf(score[gid] - m);
  ebuf[gid] = ex;
  atomicAdd(&den[d * 8 + h], ex);
}

__global__ __launch_bounds__(256)
void attn_scatter(const float* __restrict__ ebuf, const float* __restrict__ den,
                  const int* __restrict__ dstp, const __bf16* __restrict__ vb,
                  float* __restrict__ acc) {
  int e = blockIdx.x;
  int c = threadIdx.x;
  int h = c >> 5;
  int d = dstp[e];
  float alpha = ebuf[(size_t)e * 8 + h] / den[d * 8 + h];
  atomicAdd(&acc[(size_t)d * 256 + c], alpha * (float)vb[(size_t)e * 256 + c]);
}

extern "C" void kernel_launch(void* const* d_in, const int* in_sizes, int n_in,
                              void* d_out, int out_size, void* d_ws, size_t ws_size,
                              hipStream_t stream) {
  (void)in_sizes; (void)n_in; (void)out_size; (void)ws_size;
  const float* x   = (const float*)d_in[0];
  const int*   ei  = (const int*)d_in[1];
  const float* ea  = (const float*)d_in[2];
  const float* lng = (const float*)d_in[3];
  const float* lnb = (const float*)d_in[4];
  const float* qw1 = (const float*)d_in[5];  const float* qb1 = (const float*)d_in[6];
  const float* qg  = (const float*)d_in[7];  const float* qbe = (const float*)d_in[8];
  const float* qw2 = (const float*)d_in[9];  const float* qb2 = (const float*)d_in[10];
  const float* kw1 = (const float*)d_in[11]; const float* kb1 = (const float*)d_in[12];
  const float* kg  = (const float*)d_in[13]; const float* kbe = (const float*)d_in[14];
  const float* kw2 = (const float*)d_in[15]; const float* kb2 = (const float*)d_in[16];
  const float* vw1 = (const float*)d_in[17]; const float* vb1 = (const float*)d_in[18];
  const float* vg  = (const float*)d_in[19]; const float* vbe = (const float*)d_in[20];
  const float* vw2 = (const float*)d_in[21]; const float* vb2 = (const float*)d_in[22];
  const float* ow1 = (const float*)d_in[23]; const float* ob1 = (const float*)d_in[24];
  const float* ow2 = (const float*)d_in[25]; const float* ob2 = (const float*)d_in[26];
  const int* src = ei;
  const int* dst = ei + EE;

  char* ws = (char*)d_ws;
  size_t off = 0;
  auto carve = [&](size_t bytes) -> char* {
    char* p = ws + off;
    off += (bytes + 255) & ~(size_t)255;
    return p;
  };
  __bf16* wq1T = (__bf16*)carve((size_t)512 * 256 * 2);
  __bf16* wq2T = (__bf16*)carve((size_t)256 * 512 * 2);
  __bf16* wk1T = (__bf16*)carve((size_t)256 * 576 * 2);
  __bf16* wk2T = (__bf16*)carve((size_t)256 * 256 * 2);
  __bf16* wv1T = (__bf16*)carve((size_t)512 * 576 * 2);
  __bf16* wv2T = (__bf16*)carve((size_t)256 * 512 * 2);
  __bf16* wo1T = (__bf16*)carve((size_t)512 * 256 * 2);
  __bf16* wo2T = (__bf16*)carve((size_t)256 * 512 * 2);
  float*  qbuf = (float*)carve((size_t)NN * 256 * 4);
  __bf16* kbuf = (__bf16*)carve((size_t)EE * 256 * 2);
  __bf16* vbuf = (__bf16*)carve((size_t)EE * 256 * 2);
  float*  sbuf = (float*)carve((size_t)EE * 8 * 4);
  float*  ebuf = (float*)carve((size_t)EE * 8 * 4);
  int*    mxb  = (int*)carve((size_t)NN * 8 * 4);
  float*  denb = (float*)carve((size_t)NN * 8 * 4);
  float*  accb = (float*)carve((size_t)NN * 256 * 4);

  constexpr size_t SM_Q = mlp_smem(256, 512);
  constexpr size_t SM_K = mlp_smem(576, 256);
  constexpr size_t SM_V = mlp_smem(576, 512);

  hipFuncSetAttribute((const void*)&mlp_fused<256, 512, false, false, true, false>,
                      hipFuncAttributeMaxDynamicSharedMemorySize, (int)SM_Q);
  hipFuncSetAttribute((const void*)&mlp_fused<576, 256, true, true, true, true>,
                      hipFuncAttributeMaxDynamicSharedMemorySize, (int)SM_K);
  hipFuncSetAttribute((const void*)&mlp_fused<576, 512, true, true, true, true>,
                      hipFuncAttributeMaxDynamicSharedMemorySize, (int)SM_V);
  hipFuncSetAttribute((const void*)&mlp_fused<256, 512, false, false, false, false>,
                      hipFuncAttributeMaxDynamicSharedMemorySize, (int)SM_Q);

  init_attn<<<(NN * 256 + 255) / 256, 256, 0, stream>>>(mxb, denb, accb);

  auto cvt = [&](const float* w, __bf16* wT, int K, int Nc) {
    int tot = K * Nc;
    convert_wT<<<(tot + 255) / 256, 256, 0, stream>>>(w, wT, K, Nc);
  };
  cvt(qw1, wq1T, 256, 512);
  cvt(qw2, wq2T, 512, 256);
  cvt(kw1, wk1T, 576, 256);
  cvt(kw2, wk2T, 256, 256);
  cvt(vw1, wv1T, 576, 512);
  cvt(vw2, wv2T, 512, 256);
  cvt(ow1, wo1T, 256, 512);
  cvt(ow2, wo2T, 512, 256);

  // q = MLP(x): 256 -> 512 (LN,SiLU) -> 256, fp32 out
  mlp_fused<256, 512, false, false, true, false><<<(NN + 31) / 32, 256, SM_Q, stream>>>(
      x, nullptr, nullptr, nullptr, nullptr, nullptr, nullptr,
      wq1T, qb1, qg, qbe, wq2T, qb2, qbuf, NN);

  // k = MLP(LN(concat)): 576 -> 256 (LN,SiLU) -> 256, bf16 out
  mlp_fused<576, 256, true, true, true, true><<<EE / 32, 256, SM_K, stream>>>(
      nullptr, x, ea, src, dst, lng, lnb,
      wk1T, kb1, kg, kbe, wk2T, kb2, kbuf, EE);

  // v = MLP(LN(concat)): 576 -> 512 (LN,SiLU) -> 256, bf16 out
  mlp_fused<576, 512, true, true, true, true><<<EE / 32, 256, SM_V, stream>>>(
      nullptr, x, ea, src, dst, lng, lnb,
      wv1T, vb1, vg, vbe, wv2T, vb2, vbuf, EE);

  attn_score_max<<<(EE * 8 + 255) / 256, 256, 0, stream>>>(kbuf, qbuf, dst, sbuf, mxb);
  attn_exp_den<<<(EE * 8 + 255) / 256, 256, 0, stream>>>(sbuf, dst, mxb, ebuf, denb);
  attn_scatter<<<EE, 256, 0, stream>>>(ebuf, denb, dst, vbuf, accb);

  // out MLP: 256 -> 512 (SiLU only) -> 256, fp32 out to d_out
  mlp_fused<256, 512, false, false, false, false><<<(NN + 31) / 32, 256, SM_Q, stream>>>(
      accb, nullptr, nullptr, nullptr, nullptr, nullptr, nullptr,
      wo1T, ob1, nullptr, nullptr, wo2T, ob2, (float*)d_out, NN);
}
